// CausalAttention_37280316129476
// MI455X (gfx1250) — matrix operand order repaired and checked by reading, not compiled, not run on hardware
//
#include <hip/hip_runtime.h>
#include <hip/hip_bf16.h>
#include <stdint.h>

// ---------------------------------------------------------------------------
// Causal attention, B=4 S=2048 D=1024, fp32 in/out, bf16 WMMA compute.
// Double-buffered LDS tiles filled with gfx1250 async global->LDS DMA.
// ---------------------------------------------------------------------------

typedef __attribute__((ext_vector_type(16))) __bf16 v16bf;
typedef __attribute__((ext_vector_type(8)))  float  v8f;

#define BM   128
#define BN   128
#define BK   32
#define LDSP 40            // LDS row pitch in bf16 elements (80B, 16B aligned)

#define S_LEN 2048
#define D_DIM 1024
#define B_DIM 4
#define NTOK  (B_DIM * S_LEN)   // 8192

__device__ __forceinline__ unsigned short f32_to_bf16(float f) {
    union { float f; unsigned u; } in; in.f = f;
    unsigned u = in.u;
    unsigned r = u + 0x7FFFu + ((u >> 16) & 1u);   // round-to-nearest-even
    if ((u & 0x7F800000u) == 0x7F800000u) r = u;   // keep inf/nan payload
    return (unsigned short)(r >> 16);
}

union FragB16 { uint4 q[2]; v16bf v; };

// gfx1250 async DMA: 16B global -> LDS, tracked by ASYNCcnt.
__device__ __forceinline__ void async_load16(unsigned lds_off, const void* gaddr) {
    asm volatile("global_load_async_to_lds_b128 %0, %1, off"
                 :: "v"(lds_off), "v"(gaddr) : "memory");
}
__device__ __forceinline__ void wait_async0() {
    asm volatile("s_wait_asynccnt 0" ::: "memory");
}

// 128x128 output tile, K in [kBegin,kEnd) stepped by 32.
// A   : M x K row-major bf16 (lda elements)
// Bt  : N x K row-major bf16 (ldb elements)  == B stored K-contiguous per column
__device__ __forceinline__ void gemm_core(
    const unsigned short* __restrict__ A, int lda,
    const unsigned short* __restrict__ Bt, int ldb,
    int m0, int n0, int kBegin, int kEnd,
    v8f (&c)[4][2])
{
    __shared__ __align__(16) unsigned short As[2][BM][LDSP];
    __shared__ __align__(16) unsigned short Bs[2][BN][LDSP];

    const int tid  = threadIdx.x;
    const int lane = tid & 31;
    const int wave = tid >> 5;
    const int wrow = wave >> 2;        // 0..1  (64 rows each)
    const int wcol = wave & 3;         // 0..3  (32 cols each)

    const int rowA = tid >> 1;         // 0..127
    const int colA = (tid & 1) * 16;   // 0 or 16 (bf16 elements)

    v8f zero = {0.f,0.f,0.f,0.f,0.f,0.f,0.f,0.f};
    #pragma unroll
    for (int mt = 0; mt < 4; ++mt)
        #pragma unroll
        for (int nt = 0; nt < 2; ++nt)
            c[mt][nt] = zero;

    const unsigned short* gArow = A  + (size_t)(m0 + rowA) * lda + colA + kBegin;
    const unsigned short* gBrow = Bt + (size_t)(n0 + rowA) * ldb + colA + kBegin;
    const int nk = (kEnd - kBegin) / BK;

    auto issue_tile = [&](int kidx, int buf) {
        const unsigned short* ga = gArow + kidx * BK;
        const unsigned short* gb = gBrow + kidx * BK;
        const unsigned ao = (unsigned)(unsigned long long)&As[buf][rowA][colA];
        const unsigned bo = (unsigned)(unsigned long long)&Bs[buf][rowA][colA];
        async_load16(ao,      ga);
        async_load16(ao + 16, ga + 8);
        async_load16(bo,      gb);
        async_load16(bo + 16, gb + 8);
    };

    issue_tile(0, 0);
    wait_async0();
    __syncthreads();

    // A fragment: lanes 0-15 hold M=lane, K={0..7,16..23}; lanes 16-31 K={8..15,24..31}
    const int mlan = lane & 15;
    const int kaof = (lane & 16) ? 8  : 0;
    // B fragment: lanes 0-15 hold N=lane, K={0..7,16..23}; lanes 16-31 K={8..15,24..31}
    const int kbof = (lane & 16) ? 8 : 0;

    #pragma unroll 1        // keep ONE loop body so accumulators stay pinned
    for (int i = 0; i < nk; ++i) {
        const int cur = i & 1;
        const unsigned short (*Ab)[LDSP] = As[cur];
        const unsigned short (*Bb)[LDSP] = Bs[cur];
        if (i + 1 < nk) issue_tile(i + 1, cur ^ 1);   // overlap DMA with WMMA

        FragB16 b0, b1;
        {
            const int n = wcol * 32 + mlan;
            b0.q[0] = *(const uint4*)&Bb[n][kbof];
            b0.q[1] = *(const uint4*)&Bb[n][kbof + 16];
            b1.q[0] = *(const uint4*)&Bb[n + 16][kbof];
            b1.q[1] = *(const uint4*)&Bb[n + 16][kbof + 16];
        }
        #pragma unroll
        for (int mt = 0; mt < 4; ++mt) {
            FragB16 a;
            const int m = wrow * 64 + mt * 16 + mlan;
            a.q[0] = *(const uint4*)&Ab[m][kaof];
            a.q[1] = *(const uint4*)&Ab[m][kaof + 16];
            c[mt][0] = __builtin_amdgcn_wmma_f32_16x16x32_bf16(
                false, a.v, false, b0.v, (short)0, c[mt][0], false, false);
            c[mt][1] = __builtin_amdgcn_wmma_f32_16x16x32_bf16(
                false, a.v, false, b1.v, (short)0, c[mt][1], false, false);
        }

        wait_async0();        // next tile resident in the other buffer
        __syncthreads();      // single barrier per K-step (reads done before signal)
    }
}

// ---------------------------------------------------------------------------
__global__ void cast_f32_bf16_kernel(const float* __restrict__ src,
                                     unsigned short* __restrict__ dst, int n4) {
    int i = blockIdx.x * blockDim.x + threadIdx.x;
    if (i < n4) {
        float4 f = ((const float4*)src)[i];
        dst[i*4+0] = f32_to_bf16(f.x);
        dst[i*4+1] = f32_to_bf16(f.y);
        dst[i*4+2] = f32_to_bf16(f.z);
        dst[i*4+3] = f32_to_bf16(f.w);
    }
}

// z=0 -> Q (row-major), z=1 -> K (row-major), z=2 -> V stored transposed per batch
__global__ void __launch_bounds__(256) proj_kernel(
    const unsigned short* __restrict__ xb,
    const unsigned short* __restrict__ wqb,
    const unsigned short* __restrict__ wkb,
    const unsigned short* __restrict__ wvb,
    unsigned short* __restrict__ qb,
    unsigned short* __restrict__ kb,
    unsigned short* __restrict__ vtb)
{
    const int z  = blockIdx.z;
    const unsigned short* Bt = (z == 0) ? wqb : (z == 1) ? wkb : wvb;
    const int m0 = blockIdx.y * BM;
    const int n0 = blockIdx.x * BN;

    v8f c[4][2];
    gemm_core(xb, D_DIM, Bt, D_DIM, m0, n0, 0, D_DIM, c);

    const int lane = threadIdx.x & 31;
    const int wave = threadIdx.x >> 5;
    const int wrow = wave >> 2, wcol = wave & 3;
    unsigned short* outp = (z == 0) ? qb : kb;

    #pragma unroll
    for (int mt = 0; mt < 4; ++mt)
        #pragma unroll
        for (int nt = 0; nt < 2; ++nt)
            #pragma unroll
            for (int r = 0; r < 8; ++r) {
                const int gm = m0 + wrow*64 + mt*16 + r + ((lane >> 4) << 3);
                const int gn = n0 + wcol*32 + nt*16 + (lane & 15);
                const unsigned short h = f32_to_bf16(c[mt][nt][r]);
                if (z < 2) {
                    outp[(size_t)gm * D_DIM + gn] = h;
                } else {           // V^T: [batch][d][s]
                    const int bb = gm >> 11, s = gm & (S_LEN - 1);
                    vtb[((size_t)bb * D_DIM + gn) * S_LEN + s] = h;
                }
            }
}

// scores[b][q][k] = Q . K  (raw, unscaled); fully-masked tiles skipped
__global__ void __launch_bounds__(256) score_kernel(
    const unsigned short* __restrict__ qb,
    const unsigned short* __restrict__ kb,
    float* __restrict__ scores)
{
    const int b  = blockIdx.z;
    const int m0 = blockIdx.y * BM;
    const int n0 = blockIdx.x * BN;
    if (n0 > m0 + BM - 1) return;      // whole tile above the diagonal

    const unsigned short* A  = qb + (size_t)b * S_LEN * D_DIM;
    const unsigned short* Bt = kb + (size_t)b * S_LEN * D_DIM;
    v8f c[4][2];
    gemm_core(A, D_DIM, Bt, D_DIM, m0, n0, 0, D_DIM, c);

    float* Sm = scores + (size_t)b * S_LEN * S_LEN;
    const int lane = threadIdx.x & 31;
    const int wave = threadIdx.x >> 5;
    const int wrow = wave >> 2, wcol = wave & 3;
    #pragma unroll
    for (int mt = 0; mt < 4; ++mt)
        #pragma unroll
        for (int nt = 0; nt < 2; ++nt)
            #pragma unroll
            for (int r = 0; r < 8; ++r) {
                const int gm = m0 + wrow*64 + mt*16 + r + ((lane >> 4) << 3);
                const int gn = n0 + wcol*32 + nt*16 + (lane & 15);
                Sm[(size_t)gm * S_LEN + gn] = c[mt][nt][r];
            }
}

// Row softmax with 1/sqrt(1024)=1/32 scale + causal mask; bf16 probs (0 above diag)
__global__ void __launch_bounds__(256) softmax_kernel(
    const float* __restrict__ scores, unsigned short* __restrict__ probs)
{
    __shared__ float red[256];
    const int row = blockIdx.x;
    const int b = row >> 11, q = row & (S_LEN - 1);
    const float* srow = scores + ((size_t)b * S_LEN + q) * S_LEN;
    unsigned short* prow = probs + ((size_t)b * S_LEN + q) * S_LEN;
    const int tid = threadIdx.x;
    const int nv  = q + 1;
    const float kfac = 0.03125f * 1.44269504088896f;   // (1/32)*log2(e)

    float mx = -3.0e38f;
    for (int c0 = tid; c0 < nv; c0 += 256) mx = fmaxf(mx, srow[c0]);
    red[tid] = mx; __syncthreads();
    for (int s = 128; s > 0; s >>= 1) {
        if (tid < s) red[tid] = fmaxf(red[tid], red[tid + s]);
        __syncthreads();
    }
    mx = red[0]; __syncthreads();

    float sum = 0.f;
    for (int c0 = tid; c0 < nv; c0 += 256) sum += exp2f((srow[c0] - mx) * kfac);
    red[tid] = sum; __syncthreads();
    for (int s = 128; s > 0; s >>= 1) {
        if (tid < s) red[tid] += red[tid + s];
        __syncthreads();
    }
    const float inv = 1.0f / red[0];

    for (int c0 = tid; c0 < S_LEN; c0 += 256) {
        const float p = (c0 < nv) ? exp2f((srow[c0] - mx) * kfac) * inv : 0.0f;
        prow[c0] = f32_to_bf16(p);
    }
}

// out[b][q][d] = P . V   (A = probs row-major, Bt = V^T [d][s]); K clipped causally
__global__ void __launch_bounds__(256) av_kernel(
    const unsigned short* __restrict__ probs,
    const unsigned short* __restrict__ vtb,
    float* __restrict__ out)
{
    const int b  = blockIdx.z;
    const int m0 = blockIdx.y * BM;
    const int n0 = blockIdx.x * BN;
    const unsigned short* A  = probs + (size_t)b * S_LEN * S_LEN;
    const unsigned short* Bt = vtb   + (size_t)b * D_DIM * S_LEN;
    const int kEnd = m0 + BM;          // probs are zero beyond the query tile
    v8f c[4][2];
    gemm_core(A, S_LEN, Bt, S_LEN, m0, n0, 0, kEnd, c);

    float* O = out + (size_t)b * S_LEN * D_DIM;
    const int lane = threadIdx.x & 31;
    const int wave = threadIdx.x >> 5;
    const int wrow = wave >> 2, wcol = wave & 3;
    #pragma unroll
    for (int mt = 0; mt < 4; ++mt)
        #pragma unroll
        for (int nt = 0; nt < 2; ++nt)
            #pragma unroll
            for (int r = 0; r < 8; ++r) {
                const int gm = m0 + wrow*64 + mt*16 + r + ((lane >> 4) << 3);
                const int gn = n0 + wcol*32 + nt*16 + (lane & 15);
                O[(size_t)gm * D_DIM + gn] = c[mt][nt][r];
            }
}

// ---------------------------------------------------------------------------
extern "C" void kernel_launch(void* const* d_in, const int* in_sizes, int n_in,
                              void* d_out, int out_size, void* d_ws, size_t ws_size,
                              hipStream_t stream)
{
    const float* x  = (const float*)d_in[0];
    const float* wq = (const float*)d_in[1];
    const float* wk = (const float*)d_in[2];
    const float* wv = (const float*)d_in[3];
    float* out = (float*)d_out;

    char* ws = (char*)d_ws;
    unsigned short* xb  = (unsigned short*)ws; ws += (size_t)NTOK * D_DIM * 2;          // 16 MB
    unsigned short* wqb = (unsigned short*)ws; ws += (size_t)D_DIM * D_DIM * 2;         //  2 MB
    unsigned short* wkb = (unsigned short*)ws; ws += (size_t)D_DIM * D_DIM * 2;
    unsigned short* wvb = (unsigned short*)ws; ws += (size_t)D_DIM * D_DIM * 2;
    unsigned short* qb  = (unsigned short*)ws; ws += (size_t)NTOK * D_DIM * 2;          // 16 MB
    unsigned short* kb  = (unsigned short*)ws; ws += (size_t)NTOK * D_DIM * 2;          // 16 MB
    unsigned short* vtb = (unsigned short*)ws; ws += (size_t)NTOK * D_DIM * 2;          // 16 MB
    float*          sc  = (float*)ws;          ws += (size_t)B_DIM * S_LEN * S_LEN * 4; // 64 MB
    unsigned short* pb  = (unsigned short*)ws;                                           // 32 MB

    // 1) casts
    {
        const int n4x = NTOK * D_DIM / 4;
        cast_f32_bf16_kernel<<<(n4x + 255) / 256, 256, 0, stream>>>(x, xb, n4x);
        const int n4w = D_DIM * D_DIM / 4;
        cast_f32_bf16_kernel<<<(n4w + 255) / 256, 256, 0, stream>>>(wq, wqb, n4w);
        cast_f32_bf16_kernel<<<(n4w + 255) / 256, 256, 0, stream>>>(wk, wkb, n4w);
        cast_f32_bf16_kernel<<<(n4w + 255) / 256, 256, 0, stream>>>(wv, wvb, n4w);
    }
    // 2) QKV projections (z selects matrix; V stored transposed)
    proj_kernel<<<dim3(D_DIM / BN, NTOK / BM, 3), 256, 0, stream>>>(
        xb, wqb, wkb, wvb, qb, kb, vtb);
    // 3) scores = Q.K^T (causal tile skip)
    score_kernel<<<dim3(S_LEN / BN, S_LEN / BM, B_DIM), 256, 0, stream>>>(qb, kb, sc);
    // 4) softmax rows -> bf16 probs
    softmax_kernel<<<B_DIM * S_LEN, 256, 0, stream>>>(sc, pb);
    // 5) out = P.V (K-loop causally clipped)
    av_kernel<<<dim3(D_DIM / BN, S_LEN / BM, B_DIM), 256, 0, stream>>>(pb, vtb, out);
}
